// Model_19353122635871
// MI455X (gfx1250) — matrix operand order repaired and checked by reading, not compiled, not run on hardware
//
#include <hip/hip_runtime.h>

// Fused ensemble-MLP forward for MI455X (gfx1250, wave32, WMMA).
// Each wave owns 32 batch rows (two 16-row A tiles) so every B fragment
// (2x ds_load_b128) feeds two independent v_wmma_f32_16x16x32_f16 chains:
// half the LDS traffic per WMMA and two interleavable accumulator chains.

typedef __attribute__((ext_vector_type(16))) _Float16 v16h;
typedef __attribute__((ext_vector_type(8)))  _Float16 v8h;
typedef __attribute__((ext_vector_type(8)))  float    v8f;

namespace {
constexpr int NE = 8, NB = 16384, NDS = 32, NDA = 8, NDIN = 40, NH = 128;
constexpr float SLOPE = 0.01f;

// LDS layout (units: _Float16). Row strides: multiples of 8 halves (16B, so
// ds_load_b128 stays aligned), stepping 4 banks per row so the 16 per-lane
// rows of a fragment load hit distinct bank groups.
constexpr int W0_STR = 72;    // W0^T rows: K padded 40 -> 64, stride 72 (144B)
constexpr int W1_STR = 136;   // W1^T rows: K=128, stride 136 (272B)
constexpr int W2_STR = 136;   // W2^T rows: K=128
constexpr int H_STR  = 136;   // activation tile rows: 128 cols padded

constexpr int OFF_W0T = 0;                            // 128 x 72
constexpr int OFF_W1T = OFF_W0T + NH * W0_STR;        // 128 x 136
constexpr int OFF_W2T = OFF_W1T + NH * W1_STR;        // 32  x 136
constexpr int OFF_H   = OFF_W2T + NDS * W2_STR;       // 8 waves x 32 x 136
constexpr int LDS_HALVES = OFF_H + 8 * 32 * H_STR;
constexpr size_t SMEM_BYTES = (size_t)LDS_HALVES * 2; // ~128.5 KB
} // namespace

__device__ __forceinline__ v8f wmma_f16(v16h a, v16h b, v8f c) {
  // (neg_a, A, neg_b, B, c_mod, C, reuse_a, reuse_b)
  return __builtin_amdgcn_wmma_f32_16x16x32_f16(false, a, false, b, (short)0, c,
                                                false, false);
}

__device__ __forceinline__ v16h frag_cat(v8h lo, v8h hi) {
  v16h r;
#pragma unroll
  for (int t = 0; t < 8; ++t) { r[t] = lo[t]; r[t + 8] = hi[t]; }
  return r;
}

// A fragment (16x32 f16): lane (hs,row) takes K[kc+hs*8 : +8) and K[kc+16+hs*8 : +8)
__device__ __forceinline__ v16h ldsA(const _Float16* base, int row, int stride,
                                     int kc, int hs) {
  const _Float16* p = base + row * stride + kc + hs * 8;
  return frag_cat(*(const v8h*)p, *(const v8h*)(p + 16));
}

// B fragment (32x16 f16) from N-major (transposed) weights: lane with column n
// takes K[kc+hs*8 : +8) and K[kc+16+hs*8 : +8).
__device__ __forceinline__ v16h ldsB(const _Float16* base, int n, int stride,
                                     int kc, int hs) {
  const _Float16* p = base + n * stride + kc + hs * 8;
  return frag_cat(*(const v8h*)p, *(const v8h*)(p + 16));
}

__device__ __forceinline__ void cvt8(v16h& r, int off, float4 p, float4 q) {
  r[off + 0] = (_Float16)p.x; r[off + 1] = (_Float16)p.y;
  r[off + 2] = (_Float16)p.z; r[off + 3] = (_Float16)p.w;
  r[off + 4] = (_Float16)q.x; r[off + 5] = (_Float16)q.y;
  r[off + 6] = (_Float16)q.z; r[off + 7] = (_Float16)q.w;
}

// leaky_relu(v) == max(v, v*slope) for 0 < slope < 1: one mul + one max_num.
__device__ __forceinline__ float leaky(float v) { return fmaxf(v, v * SLOPE); }

// activation + f16 repack + row-major store of one 16x16 C tile
__device__ __forceinline__ void act_store(_Float16* hb, v8f c, float bias,
                                          int col, int hs) {
#pragma unroll
  for (int t = 0; t < 8; ++t) {
    float v = leaky(c[t] + bias);
    hb[(t + hs * 8) * H_STR + col] = (_Float16)v;
  }
}

// build the two layer-0 A fragments (K padded 40 -> 64) from global memory
__device__ __forceinline__ void load_inp(const float* states, const float* actions,
                                         int i, int gRow, int hs, v16h& a0, v16h& a1) {
  const float* srow = states + ((size_t)i * NB + gRow) * NDS;
  const float4* s4 = (const float4*)(srow + hs * 8);
  cvt8(a0, 0, s4[0], s4[1]);                       // K[hs*8 : +8)
  const float4* s4b = (const float4*)(srow + 16 + hs * 8);
  cvt8(a0, 8, s4b[0], s4b[1]);                     // K[16+hs*8 : +8)
#pragma unroll
  for (int t = 0; t < 16; ++t) a1[t] = (_Float16)0.0f;
  if (hs == 0) {  // K 32..39 = actions; 40..63 = zero padding
    const float4* ar = (const float4*)(actions + ((size_t)i * NB + gRow) * NDA);
    cvt8(a1, 0, ar[0], ar[1]);
  }
}

__global__ __launch_bounds__(256)
void ensemble_fused(const float* __restrict__ states, const float* __restrict__ actions,
                    const float* __restrict__ W0, const float* __restrict__ b0,
                    const float* __restrict__ W1, const float* __restrict__ b1,
                    const float* __restrict__ W2, const float* __restrict__ b2,
                    float* __restrict__ out_mean, float* __restrict__ out_var) {
  extern __shared__ char smem[];
  _Float16* sh = (_Float16*)smem;

  const int tid = threadIdx.x;
  const int j = blockIdx.y;            // model index (reduction is over inputs i)
  const int rowWG = blockIdx.x * 256;  // first batch row of this workgroup

  // ---- stage model-j weights into LDS: f32 -> f16, transposed to [n][k] ----
  {
    const float* w = W0 + (size_t)j * NDIN * NH;
#pragma unroll 1
    for (int idx = tid; idx < NH * 64; idx += 256) {
      int n = idx >> 6, k = idx & 63;
      float v = (k < NDIN) ? w[k * NH + n] : 0.0f;   // zero-pad K 40..63
      sh[OFF_W0T + n * W0_STR + k] = (_Float16)v;
    }
  }
  {
    const float* w = W1 + (size_t)j * NH * NH;
#pragma unroll 1
    for (int idx = tid; idx < NH * NH; idx += 256) {
      int n = idx >> 7, k = idx & 127;
      sh[OFF_W1T + n * W1_STR + k] = (_Float16)w[k * NH + n];
    }
  }
  {
    const float* w = W2 + (size_t)j * NH * NDS;
#pragma unroll 1
    for (int idx = tid; idx < NDS * NH; idx += 256) {
      int n = idx >> 7, k = idx & 127;
      sh[OFF_W2T + n * W2_STR + k] = (_Float16)w[k * NDS + n];
    }
  }
  __syncthreads();

  const int wave = tid >> 5;
  const int lane = tid & 31;
  const int r16 = lane & 15;   // A row within tile / C-D column within tile
  const int hs = lane >> 4;    // half-wave select
  _Float16* hbufA = sh + OFF_H + wave * (32 * H_STR);  // rows 0..15 of wave tile
  _Float16* hbufB = hbufA + 16 * H_STR;                // rows 16..31
  const int rowW = rowWG + wave * 32;                  // wave's first batch row
  const int gRowA = rowW + r16;
  const int gRowB = rowW + 16 + r16;

  const _Float16* w0t = sh + OFF_W0T;
  const _Float16* w1t = sh + OFF_W1T;
  const _Float16* w2t = sh + OFF_W2T;

  // per-lane bias scalars (column = nt*16 + r16); tiny, L2-resident
  float bias0[8], bias1[8], bias2[2];
#pragma unroll
  for (int nt = 0; nt < 8; ++nt) {
    bias0[nt] = b0[j * NH + nt * 16 + r16];
    bias1[nt] = b1[j * NH + nt * 16 + r16];
  }
#pragma unroll
  for (int nt = 0; nt < 2; ++nt) bias2[nt] = b2[j * NDS + nt * 16 + r16];

  v8f sumA[2], sqA[2], sumB[2], sqB[2];
#pragma unroll
  for (int nt = 0; nt < 2; ++nt)
#pragma unroll
    for (int t = 0; t < 8; ++t) {
      sumA[nt][t] = 0.0f; sqA[nt][t] = 0.0f;
      sumB[nt][t] = 0.0f; sqB[nt][t] = 0.0f;
    }

#pragma unroll 1
  for (int i = 0; i < NE; ++i) {
    // ---- layer-0 A fragments for both 16-row tiles ----
    v16h a0A, a1A, a0B, a1B;
    load_inp(states, actions, i, gRowA, hs, a0A, a1A);
    load_inp(states, actions, i, gRowB, hs, a0B, a1B);

    // ---- layer 0: (16x64) x (64x128); B fragments shared by both tiles ----
#pragma unroll
    for (int nt = 0; nt < 8; ++nt) {
      v16h hb[2];
#pragma unroll
      for (int kc = 0; kc < 2; ++kc)
        hb[kc] = ldsB(w0t, nt * 16 + r16, W0_STR, kc * 32, hs);
      v8f cA = {}, cB = {};
      cA = wmma_f16(a0A, hb[0], cA); cB = wmma_f16(a0B, hb[0], cB);
      cA = wmma_f16(a1A, hb[1], cA); cB = wmma_f16(a1B, hb[1], cB);
      act_store(hbufA, cA, bias0[nt], nt * 16 + r16, hs);
      act_store(hbufB, cB, bias0[nt], nt * 16 + r16, hs);
    }

    // ---- layer 1: (16x128) x (128x128) ----
    v16h haA[4], haB[4];
#pragma unroll
    for (int kc = 0; kc < 4; ++kc) {
      haA[kc] = ldsA(hbufA, r16, H_STR, kc * 32, hs);
      haB[kc] = ldsA(hbufB, r16, H_STR, kc * 32, hs);
    }
#pragma unroll
    for (int nt = 0; nt < 8; ++nt) {
      v16h hb[4];
#pragma unroll
      for (int kc = 0; kc < 4; ++kc)
        hb[kc] = ldsB(w1t, nt * 16 + r16, W1_STR, kc * 32, hs);
      v8f cA = {}, cB = {};
#pragma unroll
      for (int kc = 0; kc < 4; ++kc) {
        cA = wmma_f16(haA[kc], hb[kc], cA);
        cB = wmma_f16(haB[kc], hb[kc], cB);
      }
      act_store(hbufA, cA, bias1[nt], nt * 16 + r16, hs);
      act_store(hbufB, cB, bias1[nt], nt * 16 + r16, hs);
    }

    // ---- layer 2: (16x128) x (128x32), accumulate sum / sum^2 over i ----
#pragma unroll
    for (int kc = 0; kc < 4; ++kc) {
      haA[kc] = ldsA(hbufA, r16, H_STR, kc * 32, hs);
      haB[kc] = ldsA(hbufB, r16, H_STR, kc * 32, hs);
    }
#pragma unroll
    for (int nt = 0; nt < 2; ++nt) {
      v16h hb[4];
#pragma unroll
      for (int kc = 0; kc < 4; ++kc)
        hb[kc] = ldsB(w2t, nt * 16 + r16, W2_STR, kc * 32, hs);
      v8f cA = {}, cB = {};
#pragma unroll
      for (int kc = 0; kc < 4; ++kc) {
        cA = wmma_f16(haA[kc], hb[kc], cA);
        cB = wmma_f16(haB[kc], hb[kc], cB);
      }
      v8f pA = cA + bias2[nt];
      v8f pB = cB + bias2[nt];
      sumA[nt] += pA; sqA[nt] += pA * pA;   // FMA
      sumB[nt] += pB; sqB[nt] += pB * pB;
    }
  }

  // ---- epilogue: mean over i + states[j], unbiased variance ----
#pragma unroll
  for (int g = 0; g < 2; ++g) {
    const v8f* sum = g ? sumB : sumA;
    const v8f* sq = g ? sqB : sqA;
    const int rowG = rowW + g * 16;
#pragma unroll
    for (int nt = 0; nt < 2; ++nt) {
      v8f mean = sum[nt] * 0.125f;
      v8f var = (sq[nt] - sum[nt] * sum[nt] * 0.125f) * (1.0f / 7.0f);
      const int col = nt * 16 + r16;
#pragma unroll
      for (int t = 0; t < 8; ++t) {
        const int m = t + hs * 8;  // C/D row: VGPR t -> M = t (+8 for high half)
        const size_t o = ((size_t)j * NB + (rowG + m)) * NDS + col;
        out_mean[o] = mean[t] + states[o];
        out_var[o] = var[t];
      }
    }
  }
}

extern "C" void kernel_launch(void* const* d_in, const int* in_sizes, int n_in,
                              void* d_out, int out_size, void* d_ws, size_t ws_size,
                              hipStream_t stream) {
  (void)in_sizes; (void)n_in; (void)out_size; (void)d_ws; (void)ws_size;
  const float* states  = (const float*)d_in[0];
  const float* actions = (const float*)d_in[1];
  const float* W0 = (const float*)d_in[2];
  const float* b0 = (const float*)d_in[3];
  const float* W1 = (const float*)d_in[4];
  const float* b1 = (const float*)d_in[5];
  const float* W2 = (const float*)d_in[6];
  const float* b2 = (const float*)d_in[7];
  float* out_mean = (float*)d_out;                       // (E,B,DS)
  float* out_var  = out_mean + (size_t)NE * NB * NDS;    // (E,B,DS)

  (void)hipFuncSetAttribute((const void*)ensemble_fused,
                            hipFuncAttributeMaxDynamicSharedMemorySize,
                            (int)SMEM_BYTES);

  dim3 grid(NB / 256, NE);  // 64 batch tiles x 8 models
  ensemble_fused<<<grid, dim3(256), SMEM_BYTES, stream>>>(
      states, actions, W0, b0, W1, b1, W2, b2, out_mean, out_var);
}
